// NonlocalBlock_20031727468808
// MI455X (gfx1250) — matrix operand order repaired and checked by reading, not compiled, not run on hardware
//
#include <hip/hip_runtime.h>

typedef _Float16 h8  __attribute__((ext_vector_type(8)));
typedef _Float16 h16 __attribute__((ext_vector_type(16)));
typedef float    f8  __attribute__((ext_vector_type(8)));

#define BN_EPS 1e-5f

namespace {

constexpr int TB   = 256;   // 8 waves of 32
constexpr int BM   = 128;   // block tile M
constexpr int BNT  = 128;   // block tile N
constexpr int BK   = 32;    // K step (one wmma_f32_16x16x32_f16 per frag pair)
constexpr int AS   = 40;    // LDS row stride (halves), padded, 16B-multiple
constexpr int BS   = 40;    // LDS row stride (halves) for B^T [n][k]

// problem sizes (fixed by reference)
constexpr int C    = 256;   // input channels
constexpr int WID  = 128;   // width = C/2
constexpr int S    = 4096;  // h*w

// ---- CDNA5 async copy: global -> LDS, 16B per lane, ASYNCcnt-tracked -----
// On AMDGPU the flat address of an LDS location is {SHARED_BASE_hi, lds_off},
// so truncating a generic pointer to 32 bits yields the wave LDS byte offset.
__device__ __forceinline__ void async_b128(_Float16* lds_dst, const _Float16* gsrc) {
  unsigned int       loff = (unsigned int)(uintptr_t)lds_dst;
  unsigned long long gadr = (unsigned long long)(uintptr_t)gsrc;
  asm volatile("global_load_async_to_lds_b128 %0, %1, off"
               :: "v"(loff), "v"(gadr) : "memory");
}
__device__ __forceinline__ void wait_async0() {
  asm volatile("s_wait_asynccnt 0x0" ::: "memory");
}

// ---- fragment helper (CDNA5 ISA 7.12.2 layouts) --------------------------
__device__ __forceinline__ h16 frag_cat(const _Float16* p0, const _Float16* p1) {
  h8 lo = *(const h8*)p0;
  h8 hi = *(const h8*)p1;
  return __builtin_shufflevector(lo, hi, 0,1,2,3,4,5,6,7,8,9,10,11,12,13,14,15);
}

// ---- tile fills ----------------------------------------------------------
// Direct 2D block copy: 128 rows x 32 halves from row-major global [R][Cols]
// into LDS [row][k] stride AS. F32 path converts via VGPRs (ds_store);
// f16 path uses async-to-LDS b128 (no VGPR staging).
template<bool F32>
__device__ __forceinline__ void fill_tile(const void* G, size_t ld, int r0, int c0,
                                          _Float16* dst, int tid) {
  const int row = tid >> 1;            // 0..127
  const int cb  = (tid & 1) << 4;      // 0 or 16
  _Float16* d = dst + row * AS + cb;
  if (F32) {
    const float* src = (const float*)G + (size_t)(r0 + row) * ld + (size_t)(c0 + cb);
#pragma unroll
    for (int i = 0; i < 16; i += 4) {
      float4 f = *(const float4*)(src + i);
      d[i+0] = (_Float16)f.x; d[i+1] = (_Float16)f.y;
      d[i+2] = (_Float16)f.z; d[i+3] = (_Float16)f.w;
    }
  } else {
    const _Float16* src = (const _Float16*)G + (size_t)(r0 + row) * ld + (size_t)(c0 + cb);
    async_b128(d,     src);
    async_b128(d + 8, src + 8);
  }
}

// Stage-1 only: B = x (f32, [K=cin][N=s] row-major) -> LDS B^T [n][k] with
// f32->f16 convert + transpose (async copy cannot do either).
__device__ __forceinline__ void fill_Bt_f32(const float* Bg, size_t ldb, int k0, int n0,
                                            _Float16* sBt, int tid) {
  const int kr = tid >> 3;             // 0..31
  const int nb = (tid & 7) << 4;       // 0..112
  const float* src = Bg + (size_t)(k0 + kr) * ldb + (size_t)(n0 + nb);
#pragma unroll
  for (int i = 0; i < 16; i++) sBt[(nb + i) * BS + kr] = (_Float16)src[i];
}

// ---- WMMA GEMM core: C[128x128] = A[128xK] * B[Kx128], f32 accumulate ----
// A: row-major [M][K].  B: pre-transposed row-major [N][K] when f16 (async),
// or row-major [K][N] f32 (stage 1, transpose-convert fill).
// Double-buffered LDS: async fill of tile kt+1 overlaps WMMAs on tile kt.
template<bool AF32, bool BF32>
__device__ __forceinline__ void gemm_core(const void* Ag, size_t lda,
                                          const void* Bg, size_t ldb,
                                          int m0, int n0, int K,
                                          f8 (&acc)[4][2],
                                          _Float16 (&sA)[2][BM * AS],
                                          _Float16 (&sBt)[2][BNT * BS]) {
  const int tid  = threadIdx.x;
  const int lane = tid & 31;
  const int wave = tid >> 5;
  const int wm   = (wave & 1) << 6;    // wave M offset: 0 / 64
  const int wn   = (wave >> 1) << 5;   // wave N offset: 0 / 32 / 64 / 96
  const int lh   = lane >> 4;
  const int lm   = lane & 15;

  f8 z = {0.f, 0.f, 0.f, 0.f, 0.f, 0.f, 0.f, 0.f};
#pragma unroll
  for (int mi = 0; mi < 4; mi++) { acc[mi][0] = z; acc[mi][1] = z; }

  fill_tile<AF32>(Ag, lda, m0, 0, sA[0], tid);
  if (BF32) fill_Bt_f32((const float*)Bg, ldb, 0, n0, sBt[0], tid);
  else      fill_tile<false>(Bg, ldb, n0, 0, sBt[0], tid);
  wait_async0();
  __syncthreads();

  const int nk = K / BK;
  for (int kt = 0; kt < nk; ++kt) {
    const int cur = kt & 1, nxt = cur ^ 1;
    if (kt + 1 < nk) {
      fill_tile<AF32>(Ag, lda, m0, (kt + 1) * BK, sA[nxt], tid);
      if (BF32) fill_Bt_f32((const float*)Bg, ldb, (kt + 1) * BK, n0, sBt[nxt], tid);
      else      fill_tile<false>(Bg, ldb, n0, (kt + 1) * BK, sBt[nxt], tid);
    }

    h16 a[4], b[2];
#pragma unroll
    for (int mi = 0; mi < 4; mi++) {
      // A frag (16x32 f16): lane lm = M row; K = {8*lh+0..7, 16+8*lh+0..7}
      const _Float16* p = sA[cur] + (wm + mi * 16 + lm) * AS + (lh << 3);
      a[mi] = frag_cat(p, p + 16);
    }
#pragma unroll
    for (int ni = 0; ni < 2; ni++) {
      // B frag (32x16 f16): lane lm = N col; K = {8*lh+0..7, 16+8*lh+0..7}
      const _Float16* p = sBt[cur] + (wn + ni * 16 + lm) * BS + (lh << 3);
      b[ni] = frag_cat(p, p + 16);
    }
#pragma unroll
    for (int mi = 0; mi < 4; mi++)
#pragma unroll
      for (int ni = 0; ni < 2; ni++)
        acc[mi][ni] = __builtin_amdgcn_wmma_f32_16x16x32_f16(
            false, a[mi], false, b[ni], (short)0, acc[mi][ni], false, false);

    wait_async0();       // my wave's async writes to buf[nxt] landed
    __syncthreads();     // everyone done computing (reads of buf[cur]) + fills
  }
}

} // namespace

// ===========================================================================
// Stage 1: theta/phi/g = relu(bn(W * x))   (per batch: [128x256] x [256x4096])
//   theta -> [s][c] (A of stage 3); phi -> [c][s] (A of stage 2);
//   g -> [d][s]  (pre-transposed B of stage 2, async-copy friendly)
// ===========================================================================
__global__ void __launch_bounds__(TB) k_qkv(
    const float* __restrict__ x,
    const float* __restrict__ w_theta, const float* __restrict__ w_phi,
    const float* __restrict__ w_g,
    const float* __restrict__ g1, const float* __restrict__ b1,
    const float* __restrict__ u1, const float* __restrict__ v1,
    const float* __restrict__ g2, const float* __restrict__ b2,
    const float* __restrict__ u2, const float* __restrict__ v2,
    const float* __restrict__ g3, const float* __restrict__ b3,
    const float* __restrict__ u3, const float* __restrict__ v3,
    _Float16* __restrict__ theta_sc, _Float16* __restrict__ phi_cs,
    _Float16* __restrict__ g_ds) {
  __shared__ _Float16 sA[2][BM * AS];
  __shared__ _Float16 sBt[2][BNT * BS];

  const int batch = blockIdx.z;
  const int which = blockIdx.y;         // 0=theta 1=phi 2=g
  const int n0    = blockIdx.x * BNT;   // pixel tile

  const float* W  = which == 0 ? w_theta : (which == 1 ? w_phi : w_g);
  const float* bg = which == 0 ? g1 : (which == 1 ? g2 : g3);
  const float* bb = which == 0 ? b1 : (which == 1 ? b2 : b3);
  const float* bu = which == 0 ? u1 : (which == 1 ? u2 : u3);
  const float* bv = which == 0 ? v1 : (which == 1 ? v2 : v3);
  const float* Bx = x + (size_t)batch * C * S;

  f8 acc[4][2];
  gemm_core<true, true>(W, (size_t)C, Bx, (size_t)S, 0, n0, C, acc, sA, sBt);

  const int tid = threadIdx.x, lane = tid & 31, wave = tid >> 5;
  const int wm = (wave & 1) << 6, wn = (wave >> 1) << 5;
  const int lh = lane >> 4, lm = lane & 15;

  _Float16* dst_cs = which == 1 ? phi_cs : g_ds;     // [c][s] / [d][s]
  const size_t boff_sc = (size_t)batch * S * WID;
  const size_t boff_cs = (size_t)batch * WID * S;

#pragma unroll
  for (int mi = 0; mi < 4; mi++) {
#pragma unroll
    for (int r = 0; r < 8; r++) {
      const int m = wm + mi * 16 + (lh << 3) + r;          // out channel 0..127
      const float sc = bg[m] * rsqrtf(bv[m] + BN_EPS);
      const float sh = bb[m] - bu[m] * sc;
#pragma unroll
      for (int ni = 0; ni < 2; ni++) {
        const int s = n0 + wn + ni * 16 + lm;              // pixel
        const float val = fmaxf(acc[mi][ni][r] * sc + sh, 0.f);
        const _Float16 hv = (_Float16)val;
        if (which == 0) theta_sc[boff_sc + (size_t)s * WID + m] = hv;
        else            dst_cs[boff_cs + (size_t)m * S + s] = hv;
      }
    }
  }
}

// ===========================================================================
// Stage 2: kv[c][d] = sum_s phi[c][s] * g[d][s]^T   (128x128, K=4096) / batch
//   stored transposed: kv_t[d][c]  (pre-transposed B of stage 3)
// ===========================================================================
__global__ void __launch_bounds__(TB) k_kv(const _Float16* __restrict__ phi_cs,
                                           const _Float16* __restrict__ g_ds,
                                           _Float16* __restrict__ kv_t) {
  __shared__ _Float16 sA[2][BM * AS];
  __shared__ _Float16 sBt[2][BNT * BS];
  const int batch = blockIdx.z;

  f8 acc[4][2];
  gemm_core<false, false>(phi_cs + (size_t)batch * WID * S, (size_t)S,
                          g_ds + (size_t)batch * WID * S, (size_t)S,
                          0, 0, S, acc, sA, sBt);

  const int tid = threadIdx.x, lane = tid & 31, wave = tid >> 5;
  const int wm = (wave & 1) << 6, wn = (wave >> 1) << 5;
  const int lh = lane >> 4, lm = lane & 15;
  _Float16* out = kv_t + (size_t)batch * WID * WID;

#pragma unroll
  for (int mi = 0; mi < 4; mi++)
#pragma unroll
    for (int r = 0; r < 8; r++) {
      const int m = wm + mi * 16 + (lh << 3) + r;          // c
#pragma unroll
      for (int ni = 0; ni < 2; ni++) {
        const int n = wn + ni * 16 + lm;                   // d
        out[(size_t)n * WID + m] = (_Float16)acc[mi][ni][r];   // kv_t[d][c]
      }
    }
}

// ===========================================================================
// Stage 3: y[s][d] = sum_c theta[s][c] * kv[c][d]  (4096x128, K=128) / batch
//   Scattered directly into the reference-reshape transposed view:
//   flat = s*128+d ; c' = flat>>12 ; p = flat&4095 ; y_t[p][c'] = value
//   (y_t is the pre-transposed B of stage 4.)
// ===========================================================================
__global__ void __launch_bounds__(TB) k_attn(const _Float16* __restrict__ theta_sc,
                                             const _Float16* __restrict__ kv_t,
                                             _Float16* __restrict__ y_t) {
  __shared__ _Float16 sA[2][BM * AS];
  __shared__ _Float16 sBt[2][BNT * BS];
  const int batch = blockIdx.z;
  const int m0    = blockIdx.y * BM;     // pixel tile

  f8 acc[4][2];
  gemm_core<false, false>(theta_sc + (size_t)batch * S * WID, (size_t)WID,
                          kv_t + (size_t)batch * WID * WID, (size_t)WID,
                          m0, 0, WID, acc, sA, sBt);

  const int tid = threadIdx.x, lane = tid & 31, wave = tid >> 5;
  const int wm = (wave & 1) << 6, wn = (wave >> 1) << 5;
  const int lh = lane >> 4, lm = lane & 15;
  _Float16* out = y_t + (size_t)batch * S * WID;

#pragma unroll
  for (int mi = 0; mi < 4; mi++)
#pragma unroll
    for (int r = 0; r < 8; r++) {
      const int gm = m0 + wm + mi * 16 + (lh << 3) + r;    // pixel s
#pragma unroll
      for (int ni = 0; ni < 2; ni++) {
        const int n = wn + ni * 16 + lm;                   // channel d
        const int flat = gm * WID + n;
        const int cp = flat >> 12;                         // reshaped channel
        const int p  = flat & (S - 1);                     // reshaped pixel
        out[(size_t)p * WID + cp] = (_Float16)acc[mi][ni][r];
      }
    }
}

// ===========================================================================
// Stage 4: out = relu(bn4(w_zeta * Y) + x)
//   Y[c'][p] pre-transposed as y_t[p][c'] (ldbT=128): [256x128] x [128x4096]
// ===========================================================================
__global__ void __launch_bounds__(TB) k_zeta(
    const float* __restrict__ w_zeta, const _Float16* __restrict__ y_t,
    const float* __restrict__ x,
    const float* __restrict__ g4, const float* __restrict__ b4,
    const float* __restrict__ u4, const float* __restrict__ v4,
    float* __restrict__ out) {
  __shared__ _Float16 sA[2][BM * AS];
  __shared__ _Float16 sBt[2][BNT * BS];
  const int batch = blockIdx.z;
  const int m0    = blockIdx.y * BM;     // output channel tile (0 or 128)
  const int n0    = blockIdx.x * BNT;    // pixel tile

  f8 acc[4][2];
  gemm_core<true, false>(w_zeta, (size_t)WID,
                         y_t + (size_t)batch * S * WID, (size_t)WID,
                         m0, n0, WID, acc, sA, sBt);

  const int tid = threadIdx.x, lane = tid & 31, wave = tid >> 5;
  const int wm = (wave & 1) << 6, wn = (wave >> 1) << 5;
  const int lh = lane >> 4, lm = lane & 15;
  const size_t boff = (size_t)batch * C * S;

#pragma unroll
  for (int mi = 0; mi < 4; mi++) {
#pragma unroll
    for (int r = 0; r < 8; r++) {
      const int gm = m0 + wm + mi * 16 + (lh << 3) + r;    // out channel 0..255
      const float sc = g4[gm] * rsqrtf(v4[gm] + BN_EPS);
      const float sh = b4[gm] - u4[gm] * sc;
#pragma unroll
      for (int ni = 0; ni < 2; ni++) {
        const int gn = n0 + wn + ni * 16 + lm;             // pixel
        const size_t idx = boff + (size_t)gm * S + gn;
        out[idx] = fmaxf(acc[mi][ni][r] * sc + sh + x[idx], 0.f);
      }
    }
  }
}

// ===========================================================================
extern "C" void kernel_launch(void* const* d_in, const int* in_sizes, int n_in,
                              void* d_out, int out_size, void* d_ws, size_t ws_size,
                              hipStream_t stream) {
  const float* x       = (const float*)d_in[0];
  const float* w_theta = (const float*)d_in[1];
  const float* w_phi   = (const float*)d_in[2];
  const float* w_g     = (const float*)d_in[3];
  const float* w_zeta  = (const float*)d_in[4];
  const float* bn1g = (const float*)d_in[5],  *bn1b = (const float*)d_in[6];
  const float* bn1u = (const float*)d_in[7],  *bn1v = (const float*)d_in[8];
  const float* bn2g = (const float*)d_in[9],  *bn2b = (const float*)d_in[10];
  const float* bn2u = (const float*)d_in[11], *bn2v = (const float*)d_in[12];
  const float* bn3g = (const float*)d_in[13], *bn3b = (const float*)d_in[14];
  const float* bn3u = (const float*)d_in[15], *bn3v = (const float*)d_in[16];
  const float* bn4g = (const float*)d_in[17], *bn4b = (const float*)d_in[18];
  const float* bn4u = (const float*)d_in[19], *bn4v = (const float*)d_in[20];
  float* out = (float*)d_out;

  const int NBATCH = 16;
  const size_t nhalf = (size_t)NBATCH * S * WID;       // 8,388,608 halves / buffer
  _Float16* ws    = (_Float16*)d_ws;
  _Float16* theta = ws;                                 // [n][s][c]
  _Float16* g_ds  = theta + nhalf;                      // [n][d][s]
  _Float16* phi   = g_ds + nhalf;                       // [n][c][s]
  _Float16* y_t   = phi + nhalf;                        // [n][p][c']
  _Float16* kv_t  = y_t + nhalf;                        // [n][d][c] (0.5 MB)

  dim3 blk(TB);
  k_qkv <<<dim3(S / BNT, 3, NBATCH), blk, 0, stream>>>(
      x, w_theta, w_phi, w_g,
      bn1g, bn1b, bn1u, bn1v, bn2g, bn2b, bn2u, bn2v, bn3g, bn3b, bn3u, bn3v,
      theta, phi, g_ds);
  k_kv  <<<dim3(1, 1, NBATCH), blk, 0, stream>>>(phi, g_ds, kv_t);
  k_attn<<<dim3(1, S / BM, NBATCH), blk, 0, stream>>>(theta, kv_t, y_t);
  k_zeta<<<dim3(S / BNT, C / BM, NBATCH), blk, 0, stream>>>(
      w_zeta, y_t, x, bn4g, bn4b, bn4u, bn4v, out);
}
